// JointMambaFusion_18433999634662
// MI455X (gfx1250) — matrix operand order repaired and checked by reading, not compiled, not run on hardware
//
#include <hip/hip_runtime.h>
#include <hip/hip_bf16.h>

// ---------------------------------------------------------------------------
// JointMambaFusion for MI455X (gfx1250, wave32, WMMA).
// - All projection GEMMs on v_wmma_f32_16x16x32_f16 (f16 in, f32 accumulate),
//   16x64 output per wave (A-fragment reused across 4 column tiles).
// - Weight tiles staged in LDS, via GLOBAL_LOAD_ASYNC_TO_LDS_B128 when the
//   toolchain exposes the builtin (ASYNCcnt-tracked DMA), ds stores otherwise.
// - Selective scan is register-resident (16 SSM states per lane).
// ---------------------------------------------------------------------------

typedef __attribute__((ext_vector_type(16))) _Float16 v16h;
typedef __attribute__((ext_vector_type(8)))  float    v8f;

#if defined(__has_builtin)
#  if __has_builtin(__builtin_amdgcn_global_load_async_to_lds_b128)
#    define JMF_ASYNC_LDS 1
#  endif
#endif

#ifdef JMF_ASYNC_LDS
// builtin expects: (AS1 v4i* src, AS3 v4i* dst, imm offset, imm cpol)
typedef int jmf_v4i __attribute__((__vector_size__(16)));
typedef __attribute__((address_space(1))) jmf_v4i* jmf_gp1;
typedef __attribute__((address_space(3))) jmf_v4i* jmf_sp3;
#endif

#define B_SZ   4
#define CCH    128
#define HH     128
#define WW     128
#define DIN    256
#define LFULL  8192
#define LHALF  4096
#define MMAX   (B_SZ * LFULL)                 // 32768 rows max per direction
#define OPD    ((size_t)B_SZ * CCH * LFULL)   // per-direction outs stride (floats)

// ----------------------------------------------------------------- helpers --
__device__ __forceinline__ float silu_f(float x) { return x / (1.f + __expf(-x)); }
__device__ __forceinline__ float softplus_f(float x) {
  return (x > 20.f) ? x : log1pf(__expf(x));
}

// ------------------------------------------------------- weight conversion --
__global__ void jmf_convert_weights(const float* __restrict__ in_w,
                                    const float* __restrict__ xp_w,
                                    const float* __restrict__ ow,
                                    _Float16* __restrict__ in16,
                                    _Float16* __restrict__ xp16,
                                    _Float16* __restrict__ ow16) {
  const int NI = 4 * 512 * 128;   // 262144
  const int NX = 4 * 48 * 256;    // 49152 (x_proj padded 40 -> 48 rows)
  const int NO = 4 * 128 * 256;   // 131072
  int idx = blockIdx.x * blockDim.x + threadIdx.x;
  if (idx < NI) {
    in16[idx] = (_Float16)in_w[idx];
  } else if (idx < NI + NX) {
    int t = idx - NI;
    int k = t & 255;
    int n48 = (t >> 8) % 48;
    int dir = t / (48 * 256);
    xp16[t] = (n48 < 40) ? (_Float16)xp_w[(dir * 40 + n48) * 256 + k]
                         : (_Float16)0.f;
  } else if (idx < NI + NX + NO) {
    int t = idx - NI - NX;
    ow16[t] = (_Float16)ow[t];
  }
}

// ---------------------------------------------------------- scan gathering --
__global__ void jmf_gather_seq(const float* __restrict__ fa,
                               const float* __restrict__ fb,
                               _Float16* __restrict__ X16, int dir, int Ld) {
  int idx = blockIdx.x * blockDim.x + threadIdx.x;
  int c = idx & 127;
  int m = idx >> 7;
  int b = m / Ld;
  int l = m - b * Ld;
  int ls = (dir & 1) ? (LFULL - 1 - 2 * l) : l;   // backward: flip then ::2
  int h, w;
  const float* src;
  if (dir < 2) {                 // horizontal scan: concat along W, ::2,::2
    int r = ls >> 7, col = ls & 127;
    h = 2 * r;
    if (col < 64) { w = 2 * col;        src = fa; }
    else          { w = 2 * (col - 64); src = fb; }
  } else {                       // vertical scan: concat along H, ::2,::2, T
    int wq = ls >> 7, r2 = ls & 127;
    w = 2 * wq;
    int rr = 2 * r2;
    if (rr < 128) { h = rr;       src = fa; }
    else          { h = rr - 128; src = fb; }
  }
  float v = src[((size_t)(b * CCH + c) * HH + h) * WW + w];
  X16[(size_t)m * 128 + c] = (_Float16)v;
}

// -------------------------------------------------------------- WMMA GEMM --
// C[M, 16*NT*gridDim.y] = A[M,K] * W^T, W stored [N,K] row-major (f16).
// Block = 8 waves over 128 rows; each wave computes 16 x (16*NT) reusing its
// A-fragment across NT column tiles. Weight tile staged in LDS.
union FragH { v16h h; int4 q[2]; };

template <int K, int NT, bool TRANS>
__global__ void jmf_wmma_gemm(const _Float16* __restrict__ A,
                              const _Float16* __restrict__ W,
                              float* __restrict__ C,
                              int Nvalid, int Ld) {
  constexpr int TILE_N = 16 * NT;
  __shared__ __align__(16) _Float16 sW[TILE_N * K];
  const int tid = threadIdx.x;
  const int n0 = blockIdx.y * TILE_N;
  const bool fullTile = (n0 + TILE_N) <= Nvalid;

#ifdef JMF_ASYNC_LDS
  if (fullTile) {
    // Tile rows are contiguous in W: one contiguous DMA region, 16B per lane.
    const _Float16* gsrc = W + (size_t)n0 * K;
    for (int i = tid * 8; i < TILE_N * K; i += 256 * 8) {
      __builtin_amdgcn_global_load_async_to_lds_b128(
          (jmf_gp1)(gsrc + i), (jmf_sp3)(&sW[i]), 0, 0);
    }
    asm volatile("s_wait_asynccnt 0x0" ::: "memory");
  } else
#endif
  {
    for (int i = tid; i < TILE_N * K; i += 256) {
      int n = i / K, k = i - n * K;
      int ng = n0 + n;
      sW[i] = (ng < Nvalid) ? W[(size_t)ng * K + k] : (_Float16)0.f;
    }
  }
  __syncthreads();

  const int wave = tid >> 5;
  const int lane = tid & 31;
  const int half = lane >> 4;
  const int m0 = blockIdx.x * 128 + wave * 16;
  const _Float16* Arow = A + (size_t)(m0 + (lane & 15)) * K;
  const _Float16* Bbase = sW + (lane & 15) * K + half * 16;

  __builtin_prefetch(Arow + (size_t)128 * K, 0, 1);  // next row-block (speculative)

  v8f acc[NT] = {};
#pragma unroll
  for (int kt = 0; kt < K / 32; ++kt) {
    const int kb = kt * 32;
    FragH a;
    // A 16x32 f16 layout: elems i -> K = kb + 16*half + i
    a.q[0] = *(const int4*)(Arow + kb + half * 16);
    a.q[1] = *(const int4*)(Arow + kb + half * 16 + 8);
#pragma unroll
    for (int t = 0; t < NT; ++t) {
      FragH b;
      // B 32x16 f16 layout: lane col = lane&15, elems i -> K = kb + 16*half + i
      const _Float16* Bp = Bbase + (size_t)t * 16 * K + kb;
      b.q[0] = *(const int4*)(Bp);
      b.q[1] = *(const int4*)(Bp + 8);
      acc[t] = __builtin_amdgcn_wmma_f32_16x16x32_f16(
          false, a.h, false, b.h, (short)0, acc[t], false, false);
    }
  }

  const int Ncols = TILE_N * gridDim.y;
#pragma unroll
  for (int t = 0; t < NT; ++t) {
    const int ng = n0 + t * 16 + (lane & 15);
    if (ng < Nvalid) {
#pragma unroll
      for (int r = 0; r < 8; ++r) {
        int m = m0 + r + 8 * half;             // D layout: M = r + 8*half
        if constexpr (TRANS) {                 // store into [B, Ncols, Ld]
          int b_ = m / Ld, l = m - b_ * Ld;
          C[((size_t)(b_ * Ncols + ng)) * Ld + l] = acc[t][r];
        } else {
          C[(size_t)m * Ncols + ng] = acc[t][r];
        }
      }
    }
  }
}

// ----------------------------------------------- depthwise causal conv1d ----
__global__ void jmf_conv_silu(const float* __restrict__ xz,
                              const float* __restrict__ cw,
                              const float* __restrict__ cb,
                              float* __restrict__ xsf,
                              _Float16* __restrict__ xs16, int dir, int Ld) {
  int idx = blockIdx.x * blockDim.x + threadIdx.x;
  int d = idx & 255;
  int m = idx >> 8;
  int b = m / Ld, l = m - b * Ld;
  float acc = cb[dir * DIN + d];
#pragma unroll
  for (int k = 0; k < 4; ++k) {
    int ls = l - 3 + k;
    if (ls >= 0)
      acc += cw[(dir * DIN + d) * 4 + k] * xz[((size_t)(b * Ld + ls)) * 512 + d];
  }
  float s = silu_f(acc);
  xsf[(size_t)m * DIN + d] = s;
  xs16[(size_t)m * DIN + d] = (_Float16)s;
}

// ------------------------------------------------- dt projection + softplus -
__global__ void jmf_delta(const float* __restrict__ xdbl,
                          const float* __restrict__ dtw,
                          const float* __restrict__ dtb,
                          float* __restrict__ delta, int dir) {
  int idx = blockIdx.x * blockDim.x + threadIdx.x;
  int d = idx & 255;
  int m = idx >> 8;
  float v = dtb[dir * DIN + d];
#pragma unroll
  for (int r = 0; r < 8; ++r)
    v += xdbl[(size_t)m * 48 + r] * dtw[(dir * DIN + d) * 8 + r];
  delta[(size_t)m * DIN + d] = softplus_f(v);
}

// ------------------------------------------------------- selective scan -----
// One lane per (b, d): 16 SSM states + A row live in VGPRs; sequential over l.
// True per-step recurrence is one FMA per state; loads/exps pipeline freely.
__global__ void jmf_scan(const float* __restrict__ xsf,
                         const float* __restrict__ delta,
                         const float* __restrict__ xz,
                         const float* __restrict__ xdbl,
                         const float* __restrict__ A_log,
                         const float* __restrict__ Dp,
                         _Float16* __restrict__ ys16, int dir, int Ld) {
  const int d = threadIdx.x;
  const int b = blockIdx.x;
  float An[16], h[16];
#pragma unroll
  for (int s = 0; s < 16; ++s) {
    An[s] = -__expf(A_log[(dir * DIN + d) * 16 + s]);
    h[s] = 0.f;
  }
  const float dcoef = Dp[dir * DIN + d];
  for (int l = 0; l < Ld; ++l) {
    size_t base = (size_t)b * Ld + l;
    float u  = xsf[base * DIN + d];
    float dl = delta[base * DIN + d];
    float zv = xz[base * 512 + DIN + d];   // gate z
    // B_t (s=0..15) and C_t (s=0..15) as eight b128 loads
    union { float4 v[8]; float f[32]; } bc;
    const float4* bc4 = (const float4*)(xdbl + base * 48 + 8);
#pragma unroll
    for (int q = 0; q < 8; ++q) bc.v[q] = bc4[q];
    float du = dl * u;
    float ysum = 0.f;
#pragma unroll
    for (int s = 0; s < 16; ++s) {
      float dA = __expf(dl * An[s]);
      h[s] = dA * h[s] + du * bc.f[s];
      ysum += h[s] * bc.f[16 + s];
    }
    float y = ysum + u * dcoef;
    y *= silu_f(zv);
    ys16[base * DIN + d] = (_Float16)y;
  }
}

// ----------------------------------------------- 1D re-interp for bwd dirs --
__global__ void jmf_interp1d(const float* __restrict__ halfb,
                             float* __restrict__ full) {
  int idx = blockIdx.x * blockDim.x + threadIdx.x;
  int j = idx & (LFULL - 1);
  int bc = idx >> 13;
  float pos = 0.5f * (float)j - 0.25f;
  pos = fminf(fmaxf(pos, 0.f), (float)(LHALF - 1));
  int lo = (int)floorf(pos);
  int hi = min(lo + 1, LHALF - 1);
  float w = pos - (float)lo;
  const float* src = halfb + (size_t)bc * LHALF;
  full[(size_t)bc * LFULL + j] = src[lo] * (1.f - w) + src[hi] * w;
}

// ------------------------------------------------------------- scan merge ---
__global__ void jmf_merge(const float* __restrict__ outs,
                          float* __restrict__ a_sm, float* __restrict__ b_sm) {
  int idx = blockIdx.x * blockDim.x + threadIdx.x;
  int j = idx & 63;
  int i = (idx >> 6) & 63;
  int bc = idx >> 12;
  const float* o0 = outs + 0 * OPD + (size_t)bc * LFULL;
  const float* o1 = outs + 1 * OPD + (size_t)bc * LFULL;
  const float* o2 = outs + 2 * OPD + (size_t)bc * LFULL;
  const float* o3 = outs + 3 * OPD + (size_t)bc * LFULL;
  int lha = i * 128 + j,  lhb = lha + 64;   // feat_h columns [:64] / [64:]
  int lva = j * 128 + i,  lvb = lva + 64;   // feat_v rows    [:64] / [64:]
  a_sm[idx] = o0[lha] + o1[LFULL - 1 - lha] + o2[lva] + o3[LFULL - 1 - lva];
  b_sm[idx] = o0[lhb] + o1[LFULL - 1 - lhb] + o2[lvb] + o3[LFULL - 1 - lvb];
}

// --------------------------------------- bilinear 64->128 upsample + resid --
__device__ __forceinline__ float bilin64(const float* g, float py, float px) {
  int y0 = (int)floorf(py), x0 = (int)floorf(px);
  int y1 = min(y0 + 1, 63), x1 = min(x0 + 1, 63);
  float wy = py - (float)y0, wx = px - (float)x0;
  float v00 = g[y0 * 64 + x0], v01 = g[y0 * 64 + x1];
  float v10 = g[y1 * 64 + x0], v11 = g[y1 * 64 + x1];
  return (v00 * (1.f - wx) + v01 * wx) * (1.f - wy) +
         (v10 * (1.f - wx) + v11 * wx) * wy;
}

__global__ void jmf_upsample_add(const float* __restrict__ a_sm,
                                 const float* __restrict__ b_sm,
                                 const float* __restrict__ fa,
                                 const float* __restrict__ fb,
                                 float* __restrict__ out) {
  int idx = blockIdx.x * blockDim.x + threadIdx.x;
  int x = idx & 127;
  int y = (idx >> 7) & 127;
  int bc = idx >> 14;
  float py = fminf(fmaxf(0.5f * (float)y - 0.25f, 0.f), 63.f);
  float px = fminf(fmaxf(0.5f * (float)x - 0.25f, 0.f), 63.f);
  const float* ga = a_sm + (size_t)bc * 4096;
  const float* gb = b_sm + (size_t)bc * 4096;
  out[idx] = bilin64(ga, py, px) + bilin64(gb, py, px) + fa[idx] + fb[idx];
}

// ---------------------------------------------------------------- launcher --
extern "C" void kernel_launch(void* const* d_in, const int* in_sizes, int n_in,
                              void* d_out, int out_size, void* d_ws, size_t ws_size,
                              hipStream_t stream) {
  (void)in_sizes; (void)n_in; (void)out_size; (void)ws_size;
  const float* fa    = (const float*)d_in[0];
  const float* fb    = (const float*)d_in[1];
  const float* in_w  = (const float*)d_in[2];
  const float* cw    = (const float*)d_in[3];
  const float* cb    = (const float*)d_in[4];
  const float* xp_w  = (const float*)d_in[5];
  const float* dtw   = (const float*)d_in[6];
  const float* dtb   = (const float*)d_in[7];
  const float* A_log = (const float*)d_in[8];
  const float* Dp    = (const float*)d_in[9];
  const float* ow    = (const float*)d_in[10];
  float* out = (float*)d_out;

  // workspace layout (~276 MB total)
  char* ws = (char*)d_ws;
  size_t off = 0;
  auto carve = [&](size_t bytes) -> char* {
    char* p = ws + off;
    off = (off + bytes + 255) & ~(size_t)255;
    return p;
  };
  _Float16* in16  = (_Float16*)carve((size_t)4 * 512 * 128 * 2);
  _Float16* xp16  = (_Float16*)carve((size_t)4 * 48 * 256 * 2);
  _Float16* ow16  = (_Float16*)carve((size_t)4 * 128 * 256 * 2);
  _Float16* X16   = (_Float16*)carve((size_t)MMAX * 128 * 2);
  float*    xz    = (float*)   carve((size_t)MMAX * 512 * 4);
  float*    xsf   = (float*)   carve((size_t)MMAX * 256 * 4);
  _Float16* xs16  = (_Float16*)carve((size_t)MMAX * 256 * 2);
  float*    xdbl  = (float*)   carve((size_t)MMAX * 48 * 4);
  float*    delta = (float*)   carve((size_t)MMAX * 256 * 4);
  _Float16* ys16  = (_Float16*)carve((size_t)MMAX * 256 * 2);
  float*    outs  = (float*)   carve((size_t)4 * OPD * 4);
  float*    halfb = (float*)   carve((size_t)B_SZ * CCH * LHALF * 4);
  float*    a_sm  = (float*)   carve((size_t)B_SZ * CCH * 64 * 64 * 4);
  float*    b_sm  = (float*)   carve((size_t)B_SZ * CCH * 64 * 64 * 4);

  jmf_convert_weights<<<1728, 256, 0, stream>>>(in_w, xp_w, ow, in16, xp16, ow16);

  for (int dir = 0; dir < 4; ++dir) {
    const int Ld = (dir & 1) ? LHALF : LFULL;
    const int M = B_SZ * Ld;

    jmf_gather_seq<<<M * 128 / 256, 256, 0, stream>>>(fa, fb, X16, dir, Ld);

    // in_proj: [M,128] x [512,128]^T -> xz [M,512]   (NT=4, grid.y=8)
    jmf_wmma_gemm<128, 4, false><<<dim3(M / 128, 8), 256, 0, stream>>>(
        X16, in16 + (size_t)dir * 512 * 128, xz, 512, 0);

    jmf_conv_silu<<<M * 256 / 256, 256, 0, stream>>>(xz, cw, cb, xsf, xs16, dir, Ld);

    // x_proj: [M,256] x [40,256]^T (padded to 48) -> xdbl [M,48]  (NT=3)
    jmf_wmma_gemm<256, 3, false><<<dim3(M / 128, 1), 256, 0, stream>>>(
        xs16, xp16 + (size_t)dir * 48 * 256, xdbl, 40, 0);

    jmf_delta<<<M * 256 / 256, 256, 0, stream>>>(xdbl, dtw, dtb, delta, dir);

    jmf_scan<<<B_SZ, 256, 0, stream>>>(xsf, delta, xz, xdbl, A_log, Dp, ys16, dir, Ld);

    // out_proj: [M,256] x [128,256]^T -> transposed store into [B,128,Ld]
    float* dst = (dir & 1) ? halfb : (outs + (size_t)dir * OPD);
    jmf_wmma_gemm<256, 4, true><<<dim3(M / 128, 2), 256, 0, stream>>>(
        ys16, ow16 + (size_t)dir * 128 * 256, dst, 128, Ld);

    if (dir & 1)
      jmf_interp1d<<<B_SZ * CCH * LFULL / 256, 256, 0, stream>>>(
          halfb, outs + (size_t)dir * OPD);
  }

  jmf_merge<<<B_SZ * CCH * 64 * 64 / 256, 256, 0, stream>>>(outs, a_sm, b_sm);
  jmf_upsample_add<<<B_SZ * CCH * HH * WW / 256, 256, 0, stream>>>(
      a_sm, b_sm, fa, fb, out);
}
